// VQ_Varn_AE_73529840107861
// MI455X (gfx1250) — matrix operand order repaired and checked by reading, not compiled, not run on hardware
//
#include <hip/hip_runtime.h>

// ---------------- bf16 helpers (raw u16 storage) ----------------
__device__ __forceinline__ float bf2f(unsigned short h) {
    unsigned int u = ((unsigned int)h) << 16;
    return __builtin_bit_cast(float, u);
}
__device__ __forceinline__ unsigned short f2bf(float f) {
    unsigned int u = __builtin_bit_cast(unsigned int, f);
    u += 0x7fffu + ((u >> 16) & 1u);          // round-to-nearest-even
    return (unsigned short)(u >> 16);
}

typedef __bf16 v16bf __attribute__((ext_vector_type(16)));
typedef float  v8f   __attribute__((ext_vector_type(8)));

#define EP_NONE        0
#define EP_BIAS        1
#define EP_BIAS_LRELU  2
#define EP_BIAS_RELU   3

// B-matrix global "fragment order" index: per (nblock 64, kblock 32) tile of
// 2048 u16, laid out exactly as the per-lane WMMA B fragment (32B / lane).
__device__ __forceinline__ size_t bidx(int k, int n, int nkb) {
    int nb   = n >> 6;
    int kb   = k >> 5;
    int lane = ((n >> 4) & 3) * 32 + (n & 15) + (((k >> 4) & 1) << 4);
    return (((size_t)(nb * nkb + kb)) << 11) + lane * 16 + (k & 15);
}

// =====================================================================
// bf16 WMMA GEMM: C[M,N] = A[M,K] * B[K,N] (f32 accum)
// AMODE: 0 plain row-major A; 1 implicit conv (3x3,pad1,stride s);
//        2 implicit convT (lhs_dilation 2, pad lo=1,hi=2). NHWC bf16 act.
// CINC:  compile-time Cin (3 for conv1, per-element path); 0 = pow2 Cin,
//        vectorized 4-wide gather via shifts/masks.
// BN of the input activation folded into the A gather (identity stats when
// no BN). B is read straight from global fragment-ordered buffer (no LDS).
// Block: 256 thr = 8 waves (2M x 4N); tile 32(M) x 64(N); K step 32.
// =====================================================================
template<int AMODE, int CINC>
__global__ __launch_bounds__(256)
void gemm_bf16_wmma(const unsigned short* __restrict__ A,
                    const unsigned short* __restrict__ Bf,
                    const float* __restrict__ bias,
                    const float* __restrict__ bnst,     // [2c]=mean,[2c+1]=invstd
                    float* __restrict__ Cf,
                    unsigned short* __restrict__ Cb,
                    int M, int N, int K, int nkb,
                    int Cin, int cinsh, int Hin, int Win,
                    int Wout, int wsh, int hwsh, int stride,
                    int ldc, int coloff, int ep)
{
    __shared__ __align__(32) unsigned short Asf[1024];  // A tile, fragment order
    __shared__ float bnm[256];
    __shared__ float bni[256];

    const int tid  = threadIdx.x;
    const int lane = tid & 31;
    const int wave = tid >> 5;
    const int wM   = wave & 1;
    const int wN   = wave >> 1;
    const int m0   = blockIdx.x * 32;
    const int n0   = blockIdx.y * 64;

    if constexpr (AMODE != 0) {
        for (int i = tid; i < Cin; i += 256) {
            bnm[i] = bnst[2 * i];
            bni[i] = bnst[2 * i + 1];
        }
    }
    __syncthreads();

    // staging coords for this thread: row r, 4 consecutive k at kk0
    const int r    = tid >> 3;
    const int kk0  = (tid & 7) * 4;
    const int mg   = m0 + r;
    // swizzled LDS destination (contiguous 4 u16, 8B aligned)
    unsigned short* dst = &Asf[((r >> 4) * 32 + (r & 15) + (((kk0 >> 4) & 1) << 4)) * 16
                               + (kk0 & 15)];
    // decompose output pixel once (pow2 spatial sizes)
    const int b  = mg >> hwsh;
    const int s  = mg & ((1 << hwsh) - 1);
    const int oy = s >> wsh;
    const int ox = s & (Wout - 1);

    v8f acc = {0.f, 0.f, 0.f, 0.f, 0.f, 0.f, 0.f, 0.f};

    for (int k0 = 0; k0 < K; k0 += 32) {
        unsigned short p[4];
        if constexpr (AMODE == 0) {
            const ushort4 raw = *(const ushort4*)(A + (size_t)mg * K + (k0 + kk0));
            p[0] = raw.x; p[1] = raw.y; p[2] = raw.z; p[3] = raw.w;
        } else if constexpr (CINC > 0) {
            // per-element gather (small non-pow2 Cin, K may be < 32)
            #pragma unroll
            for (int e = 0; e < 4; ++e) {
                int kg = k0 + kk0 + e;
                float v = 0.f;
                if (kg < K) {
                    int c   = kg % CINC;
                    int tap = kg / CINC;
                    int ky = tap / 3, kx = tap - ky * 3;
                    int iy, ix; bool ok;
                    if (AMODE == 1) {
                        iy = oy * stride + ky - 1;
                        ix = ox * stride + kx - 1;
                        ok = ((unsigned)iy < (unsigned)Hin) & ((unsigned)ix < (unsigned)Win);
                    } else {
                        int ty = oy + ky - 1, tx = ox + kx - 1;
                        ok = ((unsigned)ty <= (unsigned)(2 * Hin - 2)) &
                             ((unsigned)tx <= (unsigned)(2 * Win - 2)) &
                             !(ty & 1) & !(tx & 1);
                        iy = ty >> 1; ix = tx >> 1;
                    }
                    int iyc = ok ? iy : 0, ixc = ok ? ix : 0;
                    float x = bf2f(A[(((size_t)b * Hin + iyc) * Win + ixc) * Cin + c]);
                    x = (x - bnm[c]) * bni[c];
                    v = ok ? x : 0.f;
                }
                p[e] = f2bf(v);
            }
        } else {
            // pow2 Cin >= 32: 4 consecutive k share one tap/pixel
            int kg0 = k0 + kk0;
            int c0  = kg0 & (Cin - 1);
            int tap = kg0 >> cinsh;
            int ky = tap / 3, kx = tap - ky * 3;
            int iy, ix; bool ok;
            if (AMODE == 1) {
                iy = oy * stride + ky - 1;
                ix = ox * stride + kx - 1;
                ok = ((unsigned)iy < (unsigned)Hin) & ((unsigned)ix < (unsigned)Win);
            } else {
                int ty = oy + ky - 1, tx = ox + kx - 1;
                ok = ((unsigned)ty <= (unsigned)(2 * Hin - 2)) &
                     ((unsigned)tx <= (unsigned)(2 * Win - 2)) &
                     !(ty & 1) & !(tx & 1);
                iy = ty >> 1; ix = tx >> 1;
            }
            int iyc = ok ? iy : 0, ixc = ok ? ix : 0;
            const ushort4 raw = *(const ushort4*)
                (A + (((size_t)b * Hin + iyc) * Win + ixc) * Cin + c0);
            float f0 = (bf2f(raw.x) - bnm[c0 + 0]) * bni[c0 + 0];
            float f1 = (bf2f(raw.y) - bnm[c0 + 1]) * bni[c0 + 1];
            float f2 = (bf2f(raw.z) - bnm[c0 + 2]) * bni[c0 + 2];
            float f3 = (bf2f(raw.w) - bnm[c0 + 3]) * bni[c0 + 3];
            if (!ok) { f0 = f1 = f2 = f3 = 0.f; }
            p[0] = f2bf(f0); p[1] = f2bf(f1); p[2] = f2bf(f2); p[3] = f2bf(f3);
        }
        *(ushort4*)dst = *(ushort4*)p;
        __syncthreads();

        // fragments: A from LDS (32B/lane), B straight from global (32B/lane)
        const v16bf av = *(const v16bf*)&Asf[(wM * 32 + lane) * 16];
        const v16bf bv = *(const v16bf*)
            (Bf + (((size_t)(blockIdx.y * nkb + (k0 >> 5))) << 11)
                + (size_t)((wN * 32 + lane) * 16));
        acc = __builtin_amdgcn_wmma_f32_16x16x32_bf16(
                  false, av, false, bv, (short)0, acc, false, false);
        __syncthreads();
    }

    // epilogue: C layout = 8 VGPRs, m = v + 8*(lane/16), n = lane%16
    const int ng = n0 + wN * 16 + (lane & 15);
    if (ng < N) {
        float bvv = (ep != EP_NONE && bias) ? bias[ng] : 0.f;
        #pragma unroll
        for (int v = 0; v < 8; ++v) {
            int mgo  = m0 + wM * 16 + v + 8 * (lane >> 4);
            float x = acc[v];
            if (ep != EP_NONE)       x += bvv;
            if (ep == EP_BIAS_LRELU) x = (x >= 0.f) ? x : 0.01f * x;
            if (ep == EP_BIAS_RELU)  x = (x >= 0.f) ? x : 0.f;
            size_t o = (size_t)mgo * ldc + coloff + ng;
            if (Cf) Cf[o] = x;
            else    Cb[o] = f2bf(x);
        }
    }
}

// ---------------- BatchNorm statistics over NHWC bf16 act ----------------
__global__ __launch_bounds__(256)
void bn_stats(const unsigned short* __restrict__ act, float* __restrict__ stats,
              int M, int C)
{
    int c = blockIdx.x;
    float s = 0.f, s2 = 0.f;
    for (int r = threadIdx.x; r < M; r += 256) {
        float x = bf2f(act[(size_t)r * C + c]);
        s += x; s2 += x * x;
    }
    __shared__ float sh[256], sh2[256];
    sh[threadIdx.x] = s; sh2[threadIdx.x] = s2;
    __syncthreads();
    for (int o = 128; o > 0; o >>= 1) {
        if (threadIdx.x < o) { sh[threadIdx.x] += sh[threadIdx.x + o];
                               sh2[threadIdx.x] += sh2[threadIdx.x + o]; }
        __syncthreads();
    }
    if (threadIdx.x == 0) {
        float mu  = sh[0] / (float)M;
        float var = sh2[0] / (float)M - mu * mu;
        stats[2 * c]     = mu;
        stats[2 * c + 1] = rsqrtf(var + 1e-5f);
    }
}

__global__ void init_id_stats(float* __restrict__ st)
{   // identity BN: mean 0, invstd 1 (512 floats)
    int i = blockIdx.x * 256 + threadIdx.x;
    if (i < 512) st[i] = (i & 1) ? 1.f : 0.f;
}

// ---------------- conversion / packing kernels ----------------
__global__ void convert_input(const float* __restrict__ x, unsigned short* __restrict__ xin)
{   // NCHW f32 -> NHWC bf16, [256,3,64,64]
    size_t i = (size_t)blockIdx.x * 256 + threadIdx.x;
    if (i >= (size_t)256 * 3 * 64 * 64) return;
    int xx = i & 63; size_t t = i >> 6;
    int yy = t & 63; t >>= 6;
    int c  = (int)(t % 3); int b = (int)(t / 3);
    xin[(((size_t)b * 64 + yy) * 64 + xx) * 3 + c] = f2bf(x[i]);
}

__global__ void wconv_pack(const float* __restrict__ W, unsigned short* __restrict__ Bfr,
                           int Cout, int Cin, int Kpad, int Npad, int npadsh)
{   // OIHW -> fragment order; k = (ky*3+kx)*Cin + c
    int i = blockIdx.x * 256 + threadIdx.x;
    if (i >= Kpad * Npad) return;
    int n = i & (Npad - 1);
    int k = i >> npadsh;
    int K = Cin * 9;
    float v = 0.f;
    if (n < Cout && k < K) {
        int c = k % Cin, tap = k / Cin;
        int ky = tap / 3, kx = tap % 3;
        v = W[(((size_t)n * Cin + c) * 3 + ky) * 3 + kx];
    }
    Bfr[bidx(k, n, Kpad >> 5)] = f2bf(v);
}

__global__ void wconvT_pack(const float* __restrict__ W, unsigned short* __restrict__ Bfr,
                            int Cout, int Cin, int Kpad, int Npad, int npadsh)
{   // (Cin,Cout,3,3) flipped+transposed -> fragment order
    int i = blockIdx.x * 256 + threadIdx.x;
    if (i >= Kpad * Npad) return;
    int n = i & (Npad - 1);
    int k = i >> npadsh;
    int K = Cin * 9;
    float v = 0.f;
    if (n < Cout && k < K) {
        int c = k % Cin, tap = k / Cin;
        int ky = tap / 3, kx = tap % 3;
        v = W[(((size_t)c * Cout + n) * 3 + (2 - ky)) * 3 + (2 - kx)];
    }
    Bfr[bidx(k, n, Kpad >> 5)] = f2bf(v);
}

__global__ void wfc_pack(const float* __restrict__ W, unsigned short* __restrict__ Bfr,
                         int Nout, int K, int Npad, int npadsh, int total)
{   // [Nout,K] -> fragment order, B[k][n] = W[n,k]  (also used for codebook^T)
    int i = blockIdx.x * 256 + threadIdx.x;
    if (i >= total) return;
    int n = i & (Npad - 1);
    int k = i >> npadsh;
    float v = 0.f;
    if (n < Nout) v = W[(size_t)n * K + k];     // Kpad == K here (mult of 32)
    Bfr[bidx(k, n, K >> 5)] = f2bf(v);
}

__global__ void build_fc_input(const unsigned short* __restrict__ act4,
                               const float* __restrict__ stats,
                               unsigned short* __restrict__ fcA)
{   // NHWC [B*16,256] + BN -> [B, 4096] with col = c*16 + s (NCHW flatten)
    size_t i = (size_t)blockIdx.x * 256 + threadIdx.x;
    if (i >= (size_t)256 * 4096) return;
    int s = (int)(i & 15);
    int c = (int)((i >> 4) & 255);
    int b = (int)(i >> 12);
    float x = bf2f(act4[((size_t)b * 16 + s) * 256 + c]);
    x = (x - stats[2 * c]) * stats[2 * c + 1];
    fcA[i] = f2bf(x);
}

__global__ void reshape_dec(const float* __restrict__ gfc, unsigned short* __restrict__ dact)
{   // [B,4096] (col=c*16+s) -> NHWC bf16 [B*16, 256]
    size_t i = (size_t)blockIdx.x * 256 + threadIdx.x;
    if (i >= (size_t)256 * 4096) return;
    int s = (int)(i & 15);
    int c = (int)((i >> 4) & 255);
    int b = (int)(i >> 12);
    dact[((size_t)b * 16 + s) * 256 + c] = f2bf(gfc[i]);
}

__global__ void f32_to_bf16_k(const float* __restrict__ a, unsigned short* __restrict__ b, int n)
{
    int i = blockIdx.x * 256 + threadIdx.x;
    if (i < n) b[i] = f2bf(a[i]);
}

__global__ void cb_norm(const float* __restrict__ cb, float* __restrict__ cb2, int Kcb, int D)
{
    int n = blockIdx.x * 256 + threadIdx.x;
    if (n >= Kcb) return;
    float s = 0.f;
    for (int k = 0; k < D; ++k) {
        float v = cb[(size_t)n * D + k];
        s += v * v;
    }
    cb2[n] = s;
}

__global__ __launch_bounds__(256)
void vq_argmin(const float* __restrict__ G, const float* __restrict__ cb2,
               int* __restrict__ idx, int Kcb)
{
    int m = blockIdx.x;
    float best = 3.4e38f; int bi = 0;
    for (int n = threadIdx.x; n < Kcb; n += 256) {
        float d = cb2[n] - 2.f * G[(size_t)m * Kcb + n];
        if (d < best) { best = d; bi = n; }
    }
    __shared__ float sv[256]; __shared__ int si[256];
    sv[threadIdx.x] = best; si[threadIdx.x] = bi;
    __syncthreads();
    for (int o = 128; o > 0; o >>= 1) {
        if (threadIdx.x < o) {
            float ov = sv[threadIdx.x + o]; int oi = si[threadIdx.x + o];
            if (ov < sv[threadIdx.x] ||
                (ov == sv[threadIdx.x] && oi < si[threadIdx.x])) {
                sv[threadIdx.x] = ov; si[threadIdx.x] = oi;
            }
        }
        __syncthreads();
    }
    if (threadIdx.x == 0) idx[m] = si[0];
}

__global__ __launch_bounds__(256)
void vq_gather_loss(const float* __restrict__ cb, const int* __restrict__ idx,
                    const float* __restrict__ ze, float* __restrict__ zq,
                    unsigned short* __restrict__ zqb, float* __restrict__ loss, int D)
{
    int m = blockIdx.x, d = threadIdx.x;
    float q = cb[(size_t)idx[m] * D + d];
    zq [(size_t)m * D + d] = q;
    zqb[(size_t)m * D + d] = f2bf(q);
    float diff = ze[(size_t)m * D + d] - q;
    __shared__ float sh[256];
    sh[d] = diff * diff;
    __syncthreads();
    for (int o = 128; o > 0; o >>= 1) { if (d < o) sh[d] += sh[d + o]; __syncthreads(); }
    if (d == 0) atomicAdd(loss, 2.f * sh[0]);   // emb_loss + commit_loss
}

__global__ void zero_loss(float* p) { p[0] = 0.f; }

__global__ void finalize_out(const float* __restrict__ ct4, float* __restrict__ out)
{   // [M=B*64*64, 3] -> NCHW, 0.5 + 0.5*tanh
    size_t i = (size_t)blockIdx.x * 256 + threadIdx.x;
    if (i >= (size_t)256 * 3 * 64 * 64) return;
    int xx = i & 63; size_t t = i >> 6;
    int yy = t & 63; t >>= 6;
    int c  = (int)(t % 3); int b = (int)(t / 3);
    size_t m = ((size_t)b * 64 + yy) * 64 + xx;
    out[i] = 0.5f + 0.5f * tanhf(ct4[m * 3 + c]);
}

// =====================================================================
extern "C" void kernel_launch(void* const* d_in, const int* in_sizes, int n_in,
                              void* d_out, int out_size, void* d_ws, size_t ws_size,
                              hipStream_t stream) {
    (void)in_sizes; (void)n_in; (void)out_size; (void)ws_size;
    const float* x        = (const float*)d_in[0];
    const float* enc_w1   = (const float*)d_in[1];
    const float* enc_b1   = (const float*)d_in[2];
    const float* enc_w2   = (const float*)d_in[3];
    const float* enc_b2   = (const float*)d_in[4];
    const float* enc_w3   = (const float*)d_in[5];
    const float* enc_b3   = (const float*)d_in[6];
    const float* enc_w4   = (const float*)d_in[7];
    const float* enc_b4   = (const float*)d_in[8];
    const float* enc_wmu  = (const float*)d_in[9];
    const float* enc_bmu  = (const float*)d_in[10];
    const float* enc_wcov = (const float*)d_in[11];
    const float* enc_bcov = (const float*)d_in[12];
    const float* codebook = (const float*)d_in[13];
    const float* dec_wfc  = (const float*)d_in[14];
    const float* dec_bfc  = (const float*)d_in[15];
    const float* dec_wt1  = (const float*)d_in[16];
    const float* dec_bt1  = (const float*)d_in[17];
    const float* dec_wt2  = (const float*)d_in[18];
    const float* dec_bt2  = (const float*)d_in[19];
    const float* dec_wt3  = (const float*)d_in[20];
    const float* dec_bt3  = (const float*)d_in[21];
    const float* dec_wt31 = (const float*)d_in[22];
    const float* dec_bt31 = (const float*)d_in[23];
    const float* dec_wt4  = (const float*)d_in[24];
    const float* dec_bt4  = (const float*)d_in[25];

    char* ws = (char*)d_ws;
    size_t off = 0;
    auto carve = [&](size_t bytes) -> char* {
        char* p = ws + off;
        off += (bytes + 255) & ~(size_t)255;
        return p;
    };
    unsigned short* wbuf  = (unsigned short*)carve(2097152ull * 2);   // shared B-frag buf
    unsigned short* xin   = (unsigned short*)carve(3145728ull * 2);
    unsigned short* act1  = (unsigned short*)carve(8388608ull * 2);
    unsigned short* act2  = (unsigned short*)carve(4194304ull * 2);
    unsigned short* act3  = (unsigned short*)carve(2097152ull * 2);
    unsigned short* act4  = (unsigned short*)carve(1048576ull * 2);
    float*          stats = (float*)carve(512 * 4);
    float*          idst  = (float*)carve(512 * 4);
    unsigned short* fcA   = (unsigned short*)carve(1048576ull * 2);
    float*          z_e   = (float*)carve(65536 * 4);
    unsigned short* zebf  = (unsigned short*)carve(65536 * 2);
    float*          cb2   = (float*)carve(8192 * 4);
    float*          G     = (float*)carve(2097152ull * 4);
    int*            idx   = (int*)carve(256 * 4);
    float*          zq    = (float*)carve(65536 * 4);
    unsigned short* zqbf  = (unsigned short*)carve(65536 * 2);
    float*          gfc   = (float*)carve(1048576ull * 4);
    unsigned short* dact0 = (unsigned short*)carve(1048576ull * 2);
    unsigned short* dact1 = (unsigned short*)carve(2097152ull * 2);
    unsigned short* dact2 = (unsigned short*)carve(4194304ull * 2);
    unsigned short* dact3 = (unsigned short*)carve(8388608ull * 2);
    unsigned short* dact31= (unsigned short*)carve(33554432ull * 2);
    float*          ct4   = (float*)carve(3145728ull * 4);

    float* out  = (float*)d_out;
    float* loss = out + 3145728;

    convert_input<<<12288, 256, 0, stream>>>(x, xin);
    init_id_stats<<<2, 256, 0, stream>>>(idst);
    zero_loss<<<1, 1, 0, stream>>>(loss);

    // ---- encoder ----  (args: M,N,K,nkb, Cin,cinsh,Hin,Win, Wout,wsh,hwsh,stride, ldc,coloff,ep)
    wconv_pack<<<8, 256, 0, stream>>>(enc_w1, wbuf, 32, 3, 32, 64, 6);
    gemm_bf16_wmma<1, 3><<<dim3(8192, 1), 256, 0, stream>>>(xin, wbuf, enc_b1, idst,
        nullptr, act1, 262144, 32, 27, 1, 3, 0, 64, 64, 32, 5, 10, 2, 32, 0, EP_BIAS_LRELU);
    bn_stats<<<32, 256, 0, stream>>>(act1, stats, 262144, 32);

    wconv_pack<<<72, 256, 0, stream>>>(enc_w2, wbuf, 64, 32, 288, 64, 6);
    gemm_bf16_wmma<1, 0><<<dim3(2048, 1), 256, 0, stream>>>(act1, wbuf, enc_b2, stats,
        nullptr, act2, 65536, 64, 288, 9, 32, 5, 32, 32, 16, 4, 8, 2, 64, 0, EP_BIAS_LRELU);
    bn_stats<<<64, 256, 0, stream>>>(act2, stats, 65536, 64);

    wconv_pack<<<288, 256, 0, stream>>>(enc_w3, wbuf, 128, 64, 576, 128, 7);
    gemm_bf16_wmma<1, 0><<<dim3(512, 2), 256, 0, stream>>>(act2, wbuf, enc_b3, stats,
        nullptr, act3, 16384, 128, 576, 18, 64, 6, 16, 16, 8, 3, 6, 2, 128, 0, EP_BIAS_LRELU);
    bn_stats<<<128, 256, 0, stream>>>(act3, stats, 16384, 128);

    wconv_pack<<<1152, 256, 0, stream>>>(enc_w4, wbuf, 256, 128, 1152, 256, 8);
    gemm_bf16_wmma<1, 0><<<dim3(128, 4), 256, 0, stream>>>(act3, wbuf, enc_b4, stats,
        nullptr, act4, 4096, 256, 1152, 36, 128, 7, 8, 8, 4, 2, 4, 2, 256, 0, EP_BIAS_LRELU);
    bn_stats<<<256, 256, 0, stream>>>(act4, stats, 4096, 256);

    build_fc_input<<<4096, 256, 0, stream>>>(act4, stats, fcA);

    // ---- FC heads: z_e = [mu | relu(cov)] [256,256] f32 ----
    wfc_pack<<<2048, 256, 0, stream>>>(enc_wmu, wbuf, 128, 4096, 128, 7, 524288);
    gemm_bf16_wmma<0, 0><<<dim3(8, 2), 256, 0, stream>>>(fcA, wbuf, enc_bmu, idst,
        z_e, nullptr, 256, 128, 4096, 128, 0, 0, 0, 0, 1, 0, 0, 1, 256, 0, EP_BIAS);
    wfc_pack<<<2048, 256, 0, stream>>>(enc_wcov, wbuf, 128, 4096, 128, 7, 524288);
    gemm_bf16_wmma<0, 0><<<dim3(8, 2), 256, 0, stream>>>(fcA, wbuf, enc_bcov, idst,
        z_e, nullptr, 256, 128, 4096, 128, 0, 0, 0, 0, 1, 0, 0, 1, 256, 128, EP_BIAS_RELU);

    // ---- VQ ----
    f32_to_bf16_k<<<256, 256, 0, stream>>>(z_e, zebf, 65536);
    wfc_pack<<<8192, 256, 0, stream>>>(codebook, wbuf, 8192, 256, 8192, 13, 2097152);
    cb_norm<<<32, 256, 0, stream>>>(codebook, cb2, 8192, 256);
    gemm_bf16_wmma<0, 0><<<dim3(8, 128), 256, 0, stream>>>(zebf, wbuf, nullptr, idst,
        G, nullptr, 256, 8192, 256, 8, 0, 0, 0, 0, 1, 0, 0, 1, 8192, 0, EP_NONE);
    vq_argmin<<<256, 256, 0, stream>>>(G, cb2, idx, 8192);
    vq_gather_loss<<<256, 256, 0, stream>>>(codebook, idx, z_e, zq, zqbf, loss, 256);

    // ---- decoder FC ----
    wfc_pack<<<4096, 256, 0, stream>>>(dec_wfc, wbuf, 4096, 256, 4096, 12, 1048576);
    gemm_bf16_wmma<0, 0><<<dim3(8, 64), 256, 0, stream>>>(zqbf, wbuf, dec_bfc, idst,
        gfc, nullptr, 256, 4096, 256, 8, 0, 0, 0, 0, 1, 0, 0, 1, 4096, 0, EP_BIAS);
    reshape_dec<<<4096, 256, 0, stream>>>(gfc, dact0);

    // ---- decoder transposed convs ----
    wconvT_pack<<<1152, 256, 0, stream>>>(dec_wt1, wbuf, 128, 256, 2304, 128, 7);
    gemm_bf16_wmma<2, 0><<<dim3(512, 2), 256, 0, stream>>>(dact0, wbuf, dec_bt1, idst,
        nullptr, dact1, 16384, 128, 2304, 72, 256, 8, 4, 4, 8, 3, 6, 2, 128, 0, EP_BIAS_LRELU);
    bn_stats<<<128, 256, 0, stream>>>(dact1, stats, 16384, 128);

    wconvT_pack<<<288, 256, 0, stream>>>(dec_wt2, wbuf, 64, 128, 1152, 64, 6);
    gemm_bf16_wmma<2, 0><<<dim3(2048, 1), 256, 0, stream>>>(dact1, wbuf, dec_bt2, stats,
        nullptr, dact2, 65536, 64, 1152, 36, 128, 7, 8, 8, 16, 4, 8, 2, 64, 0, EP_BIAS_LRELU);
    bn_stats<<<64, 256, 0, stream>>>(dact2, stats, 65536, 64);

    wconvT_pack<<<144, 256, 0, stream>>>(dec_wt3, wbuf, 32, 64, 576, 64, 6);
    gemm_bf16_wmma<2, 0><<<dim3(8192, 1), 256, 0, stream>>>(dact2, wbuf, dec_bt3, stats,
        nullptr, dact3, 262144, 32, 576, 18, 64, 6, 16, 16, 32, 5, 10, 2, 32, 0, EP_BIAS_LRELU);
    bn_stats<<<32, 256, 0, stream>>>(dact3, stats, 262144, 32);

    wconvT_pack<<<72, 256, 0, stream>>>(dec_wt31, wbuf, 32, 32, 288, 64, 6);
    gemm_bf16_wmma<2, 0><<<dim3(32768, 1), 256, 0, stream>>>(dact3, wbuf, dec_bt31, stats,
        nullptr, dact31, 1048576, 32, 288, 9, 32, 5, 32, 32, 64, 6, 12, 2, 32, 0, EP_BIAS_LRELU);
    bn_stats<<<32, 256, 0, stream>>>(dact31, stats, 1048576, 32);

    // final convT (stride 1, pad 1, op 0) == plain conv with flipped weights
    wconvT_pack<<<72, 256, 0, stream>>>(dec_wt4, wbuf, 3, 32, 288, 64, 6);
    gemm_bf16_wmma<1, 0><<<dim3(32768, 1), 256, 0, stream>>>(dact31, wbuf, dec_bt4, stats,
        ct4, nullptr, 1048576, 3, 288, 9, 32, 5, 64, 64, 64, 6, 12, 1, 3, 0, EP_BIAS);

    finalize_out<<<12288, 256, 0, stream>>>(ct4, out);
}
